// PIGNet_39908836114977
// MI455X (gfx1250) — compile-verified, matrix operand order checked by reading, not run on hardware
//
#include <hip/hip_runtime.h>
#include <hip/hip_bf16.h>
#include <math.h>

typedef __attribute__((ext_vector_type(16))) _Float16 v16h;
typedef __attribute__((ext_vector_type(8)))  _Float16 v8h;
typedef __attribute__((ext_vector_type(8)))  float    v8f;
typedef __attribute__((ext_vector_type(4)))  float    v4f;

#define DIM       128
#define NODES_PER 330
#define NBATCH    64

// -----------------------------------------------------------------------------
// Pack f32 weight matrix W[K][Nn] (row-major) into WMMA B-fragment layout, f16.
// Wave owning n-tile `nt` at k-tile `kt` loads, per lane, the 16 contiguous f16
// at Wp[(((nt*KT)+kt)*32 + lane)*16].
// B-matrix 16-bit layout (32x16): lane c (<16) = column c holding K 0..7 &
// 16..23; lane c+16 = column c holding K 8..15 & 24..31.
// -----------------------------------------------------------------------------
__global__ void pack_w16_kernel(const float* __restrict__ W, _Float16* __restrict__ Wp,
                                int K, int Nn) {
    int idx = blockIdx.x * blockDim.x + threadIdx.x;
    int total = K * Nn;
    if (idx >= total) return;
    int j    = idx & 15;
    int lane = (idx >> 4) & 31;
    int t    = idx >> 9;            // nt*KT + kt
    int KT   = K >> 5;
    int kt   = t % KT;
    int nt   = t / KT;
    int kh   = (lane >> 4) * 8;     // 0 or 8
    int koff = (j < 8) ? (kh + j) : (16 + kh + (j - 8));
    int k    = kt * 32 + koff;
    int col  = nt * 16 + (lane & 15);
    Wp[idx] = (_Float16)W[(size_t)k * Nn + col];
}

// -----------------------------------------------------------------------------
// GNN GEMM: out[M][128] = A[M][128] @ W  (W prepacked, f16 fragments)
// One workgroup = 32 rows (2 M-tiles), 8 waves = 8 n-tiles of 16 cols.
// K = 128 (4 k-tiles): each wave issues 8 WMMAs per B-fragment stream.
// -----------------------------------------------------------------------------
__global__ void gemm_gnn_kernel(const float* __restrict__ A,
                                const _Float16* __restrict__ Wp,
                                float* __restrict__ out, int M) {
    __shared__ _Float16 xa[32 * 144];   // 32 rows, stride 144 f16 (288B, 16B-aligned)
    const int tid  = threadIdx.x;
    const int row0 = blockIdx.x * 32;
    {   // cooperative load + f32->f16 convert: 256 threads x 16 values
        int r  = tid >> 3;            // 0..31
        int c0 = (tid & 7) * 16;      // 0..112
        int rr = row0 + r; if (rr >= M) rr = M - 1;
        const float* sp = A + (size_t)rr * DIM + c0;
        _Float16* dp = xa + r * 144 + c0;
        #pragma unroll
        for (int i = 0; i < 16; ++i) dp[i] = (_Float16)sp[i];
    }
    __syncthreads();
    const int lane = tid & 31;
    const int wave = tid >> 5;          // n-tile 0..7
    const int ar   = lane & 15;
    const int kh   = (lane >> 4) * 8;
    v8f acc0 = {}, acc1 = {};
    #pragma unroll
    for (int kt = 0; kt < 4; ++kt) {
        const int kb = kt * 32;
        v16h a0, a1;
        {
            v8h lo = *(const v8h*)&xa[ar * 144 + kb + kh];
            v8h hi = *(const v8h*)&xa[ar * 144 + kb + 16 + kh];
            #pragma unroll
            for (int i = 0; i < 8; ++i) { a0[i] = lo[i]; a0[8 + i] = hi[i]; }
        }
        {
            v8h lo = *(const v8h*)&xa[(16 + ar) * 144 + kb + kh];
            v8h hi = *(const v8h*)&xa[(16 + ar) * 144 + kb + 16 + kh];
            #pragma unroll
            for (int i = 0; i < 8; ++i) { a1[i] = lo[i]; a1[8 + i] = hi[i]; }
        }
        v16h b = *(const v16h*)&Wp[(size_t)((wave * 4 + kt) * 32 + lane) * 16];
        acc0 = __builtin_amdgcn_wmma_f32_16x16x32_f16(false, a0, false, b,
                                                      (short)0, acc0, false, false);
        acc1 = __builtin_amdgcn_wmma_f32_16x16x32_f16(false, a1, false, b,
                                                      (short)0, acc1, false, false);
    }
    const int rbase = (lane >> 4) << 3;   // 0 or 8
    const int col   = wave * 16 + ar;
    if (row0 + 32 <= M) {                 // uniform fast path, no per-element guards
        #pragma unroll
        for (int v = 0; v < 8; ++v) {
            out[(size_t)(row0 + rbase + v) * DIM + col]      = acc0[v];
            out[(size_t)(row0 + 16 + rbase + v) * DIM + col] = acc1[v];
        }
    } else {
        #pragma unroll
        for (int v = 0; v < 8; ++v) {
            int r0 = row0 + rbase + v, r1 = row0 + 16 + rbase + v;
            if (r0 < M) out[(size_t)r0 * DIM + col] = acc0[v];
            if (r1 < M) out[(size_t)r1 * DIM + col] = acc1[v];
        }
    }
}

// msg scatter-add: accum[dst] += msg[src], 128 floats/edge, 32 lanes x 4 floats
__global__ void scatter_add_kernel(const int* __restrict__ ei, int E,
                                   const float* __restrict__ msg,
                                   float* __restrict__ accum) {
    long long idx = (long long)blockIdx.x * blockDim.x + threadIdx.x;
    if (idx >= (long long)E * 32) return;
    int lane = (int)(idx & 31);
    int e    = (int)(idx >> 5);
    int s = ei[e];
    int d = ei[E + e];
    const float* mp = msg + (size_t)s * DIM + lane * 4;
    float* ap = accum + (size_t)d * DIM + lane * 4;
    #pragma unroll
    for (int i = 0; i < 4; ++i) atomicAdd(&ap[i], mp[i]);
}

__global__ void relu_kernel(const float* __restrict__ in, float* __restrict__ out, int n) {
    int i = blockIdx.x * blockDim.x + threadIdx.x;
    if (i < n) { float v = in[i]; out[i] = v > 0.f ? v : 0.f; }
}

__global__ void zero_kernel(float* __restrict__ p, int n) {
    int i = blockIdx.x * blockDim.x + threadIdx.x;
    if (i < n) p[i] = 0.f;
}

// -----------------------------------------------------------------------------
// Interaction kernel: 32 edges per workgroup (2 M-tiles). Fused gather +
// dual MLP (WMMA) + bias/relu/2nd-layer dot + physics + per-batch energies.
// The f16 A-tile LDS buffer (32x272 f16 = 17408B) is reused after the GEMM as
// two f32 staging planes (2 x 32 x 136 f32 = 34816B) for a conflict-free
// second-layer reduction.
// -----------------------------------------------------------------------------
__global__ void interact_kernel(const float* __restrict__ x,
                                const float* __restrict__ pos,
                                const float* __restrict__ vdw_radii,
                                const unsigned char* __restrict__ is_metal,
                                const unsigned char* __restrict__ is_don,
                                const unsigned char* __restrict__ is_acc,
                                const unsigned char* __restrict__ is_hyd,
                                const int* __restrict__ ei, int E,
                                const _Float16* __restrict__ Wp_eps,
                                const float* __restrict__ b_eps1,
                                const float* __restrict__ W_eps2,
                                const float* __restrict__ b_eps2,
                                const _Float16* __restrict__ Wp_dv,
                                const float* __restrict__ b_dv1,
                                const float* __restrict__ W_dv2,
                                const float* __restrict__ b_dv2,
                                const float* __restrict__ hbond_c,
                                const float* __restrict__ hydro_c,
                                float* __restrict__ raw_energy,
                                float* __restrict__ dvdw_out) {
    __shared__ float smem[8704];        // aliased: f16 A-tile OR 2 f32 staging planes
    __shared__ int   s_src[32], s_dst[32];
    __shared__ float red_eps[32], red_dv[32];
    _Float16* xa = (_Float16*)smem;     // [32][272] f16

    const int tid = threadIdx.x;
    const int e0  = blockIdx.x * 32;
    if (tid < 32) {
        int e = e0 + tid; if (e >= E) e = E - 1;
        s_src[tid] = ei[e];
        s_dst[tid] = ei[E + e];
        red_eps[tid] = 0.f;
        red_dv[tid]  = 0.f;
    }
    __syncthreads();
    {   // gather x_cat tile: 32 edges x 256 cols (src 0..127, dst 128..255)
        int r    = tid >> 3;            // edge 0..31
        int c0   = (tid & 7) * 32;      // 0,32,...,224  (each segment within one half)
        int node = (c0 < DIM) ? s_src[r] : s_dst[r];
        const float* sp = x + (size_t)node * DIM + (c0 & (DIM - 1));
        _Float16* dp = xa + r * 272 + c0;
        #pragma unroll
        for (int i = 0; i < 32; ++i) dp[i] = (_Float16)sp[i];
    }
    __syncthreads();

    const int lane = tid & 31;
    const int wave = tid >> 5;
    const int ar   = lane & 15;
    const int kh   = (lane >> 4) * 8;
    v8f acc_e0 = {}, acc_e1 = {}, acc_d0 = {}, acc_d1 = {};
    #pragma unroll
    for (int kt = 0; kt < 8; ++kt) {
        const int kb = kt * 32;
        v16h a0, a1;
        {
            v8h lo = *(const v8h*)&xa[ar * 272 + kb + kh];
            v8h hi = *(const v8h*)&xa[ar * 272 + kb + 16 + kh];
            #pragma unroll
            for (int i = 0; i < 8; ++i) { a0[i] = lo[i]; a0[8 + i] = hi[i]; }
        }
        {
            v8h lo = *(const v8h*)&xa[(16 + ar) * 272 + kb + kh];
            v8h hi = *(const v8h*)&xa[(16 + ar) * 272 + kb + 16 + kh];
            #pragma unroll
            for (int i = 0; i < 8; ++i) { a1[i] = lo[i]; a1[8 + i] = hi[i]; }
        }
        const size_t bo = (size_t)((wave * 8 + kt) * 32 + lane) * 16;
        v16h be = *(const v16h*)&Wp_eps[bo];
        v16h bd = *(const v16h*)&Wp_dv[bo];
        if (kt < 7) {   // prefetch next k-tile B fragments (global_prefetch_b8)
            __builtin_prefetch((const void*)&Wp_eps[bo + 512], 0, 0);
            __builtin_prefetch((const void*)&Wp_dv[bo + 512], 0, 0);
        }
        acc_e0 = __builtin_amdgcn_wmma_f32_16x16x32_f16(false, a0, false, be,
                                                        (short)0, acc_e0, false, false);
        acc_e1 = __builtin_amdgcn_wmma_f32_16x16x32_f16(false, a1, false, be,
                                                        (short)0, acc_e1, false, false);
        acc_d0 = __builtin_amdgcn_wmma_f32_16x16x32_f16(false, a0, false, bd,
                                                        (short)0, acc_d0, false, false);
        acc_d1 = __builtin_amdgcn_wmma_f32_16x16x32_f16(false, a1, false, bd,
                                                        (short)0, acc_d1, false, false);
    }
    __syncthreads();    // all A-tile reads done; reuse smem as f32 staging

    // Stage relu(h + b1) * w2 products (conflict-free b32 stores).
    float* stg_e = smem;                  // [32][136]
    float* stg_d = smem + 32 * 136;       // [32][136]
    {
        const int   col = wave * 16 + ar;
        const float be1 = b_eps1[col], bd1 = b_dv1[col];
        const float we2 = W_eps2[col], wd2 = W_dv2[col];
        const int rbase = (lane >> 4) << 3;
        #pragma unroll
        for (int v = 0; v < 8; ++v) {
            int r0 = rbase + v, r1 = 16 + rbase + v;
            float he0 = acc_e0[v] + be1; he0 = he0 > 0.f ? he0 : 0.f;
            float he1 = acc_e1[v] + be1; he1 = he1 > 0.f ? he1 : 0.f;
            float hd0 = acc_d0[v] + bd1; hd0 = hd0 > 0.f ? hd0 : 0.f;
            float hd1 = acc_d1[v] + bd1; hd1 = hd1 > 0.f ? hd1 : 0.f;
            stg_e[r0 * 136 + col] = he0 * we2;
            stg_e[r1 * 136 + col] = he1 * we2;
            stg_d[r0 * 136 + col] = hd0 * wd2;
            stg_d[r1 * 136 + col] = hd1 * wd2;
        }
    }
    __syncthreads();

    // Reduce 128 columns per edge: 8 threads/edge, vectorized LDS reads.
    {
        int r   = tid >> 3;               // edge 0..31
        int c0  = (tid & 7) * 16;         // 16-col segment
        const v4f* pe = (const v4f*)&stg_e[r * 136 + c0];
        const v4f* pd = (const v4f*)&stg_d[r * 136 + c0];
        float se = 0.f, sd = 0.f;
        #pragma unroll
        for (int q = 0; q < 4; ++q) {
            v4f ve = pe[q], vd = pd[q];
            se += ve.x + ve.y + ve.z + ve.w;
            sd += vd.x + vd.y + vd.z + vd.w;
        }
        atomicAdd(&red_eps[r], se);
        atomicAdd(&red_dv[r],  sd);
    }
    __syncthreads();

    if (tid < 32 && (e0 + tid) < E) {
        const int e = e0 + tid;
        const int s = s_src[tid], d = s_dst[tid];
        float dx = pos[s * 3 + 0] - pos[d * 3 + 0];
        float dy = pos[s * 3 + 1] - pos[d * 3 + 1];
        float dz = pos[s * 3 + 2] - pos[d * 3 + 2];
        float D  = sqrtf(dx * dx + dy * dy + dz * dz + 1e-12f);
        float inr = (D >= 0.5f && D <= 5.0f) ? 1.f : 0.f;

        float dv  = tanhf(red_dv[tid] + b_dv2[0]) * 0.2f;     // DEV_VDW_COEFF
        float R   = vdw_radii[s] + vdw_radii[d] + dv;
        float eps = 1.f / (1.f + expf(-(red_eps[tid] + b_eps2[0])));
        eps = eps * (0.0356f - 0.0178f) + 0.0178f;

        float dm = R / D;
        float n  = (D < R) ? 10.f : 6.f;
        float p  = powf(dm, n);
        float e_vdw = eps * (p * p - 2.f * p);

        float hb = hbond_c[0], hp = hydro_c[0];
        float min_hb = -hb * hb, min_hp = -hp * hp;
        float e_hb = min_hb * fminf(fmaxf((R - D) / 0.7f, 0.f), 1.f);       // cuts (-0.7, 0)
        float e_hp = min_hp * fminf(fmaxf(1.5f - (D - R), 0.f), 1.f);       // cuts (0.5, 1.5)

        bool mi = is_metal[s] != 0, mj = is_metal[d] != 0;
        bool di = is_don[s]   != 0, dj = is_don[d]   != 0;
        bool ai = is_acc[s]   != 0, aj = is_acc[d]   != 0;
        bool hi = is_hyd[s]   != 0, hj = is_hyd[d]   != 0;
        bool nom = !(mi || mj);
        float m_vdw = nom ? 1.f : 0.f;
        float m_hb  = (((di && aj) || (ai && dj)) && nom) ? 1.f : 0.f;
        float m_ml  = ((mi && aj) || (ai && mj)) ? 1.f : 0.f;
        float m_hp  = (hi && hj && nom) ? 1.f : 0.f;

        int b = s / NODES_PER;
        atomicAdd(&raw_energy[b * 4 + 0], e_vdw * m_vdw * inr);
        atomicAdd(&raw_energy[b * 4 + 1], e_hb  * m_hb  * inr);
        atomicAdd(&raw_energy[b * 4 + 2], e_hb  * m_ml  * inr);
        atomicAdd(&raw_energy[b * 4 + 3], e_hp  * m_hp  * inr);
        dvdw_out[e] = dv * inr;
    }
}

__global__ void finalize_kernel(const float* __restrict__ raw,
                                const float* __restrict__ rotor,
                                const float* __restrict__ rotor_c,
                                float* __restrict__ out) {
    int i = blockIdx.x * blockDim.x + threadIdx.x;
    if (i >= NBATCH * 4) return;
    float rc = rotor_c[0];
    out[i] = raw[i] / (1.f + rc * rc * rotor[i >> 2]);
}

// -----------------------------------------------------------------------------
extern "C" void kernel_launch(void* const* d_in, const int* in_sizes, int n_in,
                              void* d_out, int out_size, void* d_ws, size_t ws_size,
                              hipStream_t stream) {
    const float* x_in        = (const float*)d_in[0];
    const float* pos         = (const float*)d_in[1];
    const float* vdw_radii   = (const float*)d_in[2];
    const float* rotor       = (const float*)d_in[3];
    const unsigned char* is_metal = (const unsigned char*)d_in[4];
    const unsigned char* is_don   = (const unsigned char*)d_in[5];
    const unsigned char* is_acc   = (const unsigned char*)d_in[6];
    const unsigned char* is_hyd   = (const unsigned char*)d_in[7];
    const int* ei  = (const int*)d_in[8];
    const int* eic = (const int*)d_in[9];
    const int* eii = (const int*)d_in[10];
    const float* W_self_intra = (const float*)d_in[12];
    const float* W_msg_intra  = (const float*)d_in[13];
    const float* W_self_inter = (const float*)d_in[14];
    const float* W_msg_inter  = (const float*)d_in[15];
    const float* W_eps1  = (const float*)d_in[16];
    const float* b_eps1  = (const float*)d_in[17];
    const float* W_eps2  = (const float*)d_in[18];
    const float* b_eps2  = (const float*)d_in[19];
    const float* W_dv1   = (const float*)d_in[20];
    const float* b_dv1   = (const float*)d_in[21];
    const float* W_dv2   = (const float*)d_in[22];
    const float* b_dv2   = (const float*)d_in[23];
    const float* hbond_c = (const float*)d_in[24];
    const float* hydro_c = (const float*)d_in[25];
    const float* rotor_c = (const float*)d_in[26];

    const int N  = in_sizes[0] / DIM;    // 21120
    const int E1 = in_sizes[8] / 2;      // 168960
    const int Ec = in_sizes[9] / 2;      // 256000
    const int Ei = in_sizes[10] / 2;     // 576000

    auto aup = [](size_t v) { return (v + 255) & ~(size_t)255; };
    char* ws = (char*)d_ws;
    size_t off = 0;
    const size_t nodeBytes = (size_t)N * DIM * sizeof(float);
    float* xbuf = (float*)(ws + off); off += aup(nodeBytes);
    float* bufS = (float*)(ws + off); off += aup(nodeBytes);
    float* bufM = (float*)(ws + off); off += aup(nodeBytes);
    _Float16* WpG = (_Float16*)(ws + off); off += aup((size_t)12 * DIM * DIM * 2);
    _Float16* WpE = (_Float16*)(ws + off); off += aup((size_t)2 * DIM * DIM * 2);
    _Float16* WpD = (_Float16*)(ws + off); off += aup((size_t)2 * DIM * DIM * 2);
    float* raw = (float*)(ws + off); off += aup(NBATCH * 4 * sizeof(float));

    // ---- pack all weight matrices to f16 WMMA fragment layout ----
    const int mat = DIM * DIM;            // 16384
    for (int l = 0; l < 3; ++l) {
        pack_w16_kernel<<<mat / 256, 256, 0, stream>>>(W_self_intra + (size_t)l * mat, WpG + (size_t)(0 + l) * mat, DIM, DIM);
        pack_w16_kernel<<<mat / 256, 256, 0, stream>>>(W_msg_intra  + (size_t)l * mat, WpG + (size_t)(3 + l) * mat, DIM, DIM);
        pack_w16_kernel<<<mat / 256, 256, 0, stream>>>(W_self_inter + (size_t)l * mat, WpG + (size_t)(6 + l) * mat, DIM, DIM);
        pack_w16_kernel<<<mat / 256, 256, 0, stream>>>(W_msg_inter  + (size_t)l * mat, WpG + (size_t)(9 + l) * mat, DIM, DIM);
    }
    pack_w16_kernel<<<(2 * mat) / 256, 256, 0, stream>>>(W_eps1, WpE, 2 * DIM, DIM);
    pack_w16_kernel<<<(2 * mat) / 256, 256, 0, stream>>>(W_dv1,  WpD, 2 * DIM, DIM);

    // ---- GNN layers ----
    const int mtiles    = (N + 31) / 32;
    const int reluN     = N * DIM;
    const int reluGrid  = (reluN + 255) / 256;
    const float* cur = x_in;
    for (int l = 0; l < 3; ++l) {   // intra
        gemm_gnn_kernel<<<mtiles, 256, 0, stream>>>(cur, WpG + (size_t)(0 + l) * mat, bufS, N);
        gemm_gnn_kernel<<<mtiles, 256, 0, stream>>>(cur, WpG + (size_t)(3 + l) * mat, bufM, N);
        long long st = (long long)E1 * 32;
        scatter_add_kernel<<<(unsigned)((st + 255) / 256), 256, 0, stream>>>(ei, E1, bufM, bufS);
        relu_kernel<<<reluGrid, 256, 0, stream>>>(bufS, xbuf, reluN);
        cur = xbuf;
    }
    for (int l = 0; l < 3; ++l) {   // inter (cross edges)
        gemm_gnn_kernel<<<mtiles, 256, 0, stream>>>(cur, WpG + (size_t)(6 + l) * mat, bufS, N);
        gemm_gnn_kernel<<<mtiles, 256, 0, stream>>>(cur, WpG + (size_t)(9 + l) * mat, bufM, N);
        long long st = (long long)Ec * 32;
        scatter_add_kernel<<<(unsigned)((st + 255) / 256), 256, 0, stream>>>(eic, Ec, bufM, bufS);
        relu_kernel<<<reluGrid, 256, 0, stream>>>(bufS, xbuf, reluN);
        cur = xbuf;
    }

    // ---- interaction energies ----
    zero_kernel<<<1, 256, 0, stream>>>(raw, NBATCH * 4);
    float* out_f     = (float*)d_out;
    float* dvdw_out  = out_f + NBATCH * 4;
    interact_kernel<<<(Ei + 31) / 32, 256, 0, stream>>>(
        xbuf, pos, vdw_radii, is_metal, is_don, is_acc, is_hyd,
        eii, Ei, WpE, b_eps1, W_eps2, b_eps2, WpD, b_dv1, W_dv2, b_dv2,
        hbond_c, hydro_c, raw, dvdw_out);
    finalize_kernel<<<1, 256, 0, stream>>>(raw, rotor, rotor_c, out_f);
}
